// CrossGraphMessagePassing_53979148976478
// MI455X (gfx1250) — matrix operand order repaired and checked by reading, not compiled, not run on hardware
//
#include <hip/hip_runtime.h>

#define NP 40000
#define NL 10000
#define NE 1000000
#define INDIM 276
#define HID 128
#define NTILES (NE/16)

typedef __attribute__((ext_vector_type(16))) _Float16 v16h;
typedef __attribute__((ext_vector_type(8)))  float    v8f;

// workspace byte offsets (all 128B aligned)
#define OFF_LOGITS 0u
#define OFF_SEGMAX 4000000u
#define OFF_SEGSUM 4040960u
#define OFF_AGG    4081920u
#define OFF_W1A    9201920u   // att_w1 fragments: 9*8*512 halves
#define OFF_W1P    9275648u   // proj_w1 fragments
#define OFF_W2P    9349376u   // proj_w2 fragments: 4*8*512 halves

// Wave-local LDS fence: LDS ops from one wave complete in-order (ISA 7.1);
// the asm blocks compiler reordering and conservatively drains DScnt.
#define WAVE_LDS_SYNC() asm volatile("s_wait_dscnt 0" ::: "memory")

__device__ __forceinline__ float silu_f(float x){ return x * (1.0f/(1.0f+__expf(-x))); }

__device__ __forceinline__ float unmap_max(unsigned u){
  unsigned f = (u & 0x80000000u) ? (u ^ 0x80000000u) : ~u;
  return __uint_as_float(f);
}

// A fragment for V_WMMA_F32_16X16X32_F16 from a row-major f16 tile in LDS.
// Lane l holds row m=l&15; lanes>=16 hold the K+8 / K+24 halves (ISA 7.12.2).
__device__ __forceinline__ v16h load_a_frag(const _Float16* base, int lane, int kbase, int rowStride){
  int m  = lane & 15;
  int kh = (lane >> 4) & 1;
  const _Float16* row = base + m*rowStride;
  union { v16h h; unsigned u[8]; } fr;
  int b0 = kbase + 8*kh;
#pragma unroll
  for (int j=0;j<8;j++){
    int k = b0 + 2*j + ((j>=4)?8:0);
    fr.u[j] = *(const unsigned*)(row + k);   // two consecutive halves -> one VGPR
  }
  return fr.h;
}

__device__ __forceinline__ void stage_geom(int e0, int lane,
    const int* pidx, const int* lidx,
    const float* ppos, const float* lpos,
    _Float16* catW, int* pidW, int* lidW, float* dist_out)
{
  if (lane < 16){
    int e = e0 + lane;
    int p = pidx[e]; int l = lidx[e];
    pidW[lane] = p; lidW[lane] = l;
    float dx = lpos[l*3+0]-ppos[p*3+0];
    float dy = lpos[l*3+1]-ppos[p*3+1];
    float dz = lpos[l*3+2]-ppos[p*3+2];
    float d  = sqrtf(dx*dx+dy*dy+dz*dz);
    *dist_out = d;
    float inv = 1.0f/(d+1e-8f);
    _Float16* row = catW + lane*288;
    row[256]=(_Float16)d;
    row[257]=(_Float16)(dx*inv);
    row[258]=(_Float16)(dy*inv);
    row[259]=(_Float16)(dz*inv);
#pragma unroll
    for (int i=0;i<16;i++){
      float c = (8.0f/15.0f)*(float)i;      // linspace(0,8,16)
      float t = d - c;
      row[260+i]=(_Float16)__expf(-2.0f*t*t); // gamma = 2
    }
#pragma unroll
    for (int i=276;i<288;i++) row[i]=(_Float16)0.0f; // K padding
  }
}

__device__ __forceinline__ void stage_feat(int lane,
    const float* h_prot, const float* h_lig,
    const int* pidW, const int* lidW, _Float16* catW)
{
#pragma unroll 4
  for (int m=0;m<16;m++){
    const float4* hp = (const float4*)(h_prot + (size_t)pidW[m]*128);
    const float4* hl = (const float4*)(h_lig  + (size_t)lidW[m]*128);
    float4 a = hp[lane];                       // b128 load: 4 channels/lane
    float4 b = hl[lane];
    union{ uint2 u; _Float16 h[4]; } pa, pb;
    pa.h[0]=(_Float16)a.x; pa.h[1]=(_Float16)a.y; pa.h[2]=(_Float16)a.z; pa.h[3]=(_Float16)a.w;
    pb.h[0]=(_Float16)b.x; pb.h[1]=(_Float16)b.y; pb.h[2]=(_Float16)b.z; pb.h[3]=(_Float16)b.w;
    uint2* rowq = (uint2*)(catW + m*288);
    rowq[lane]    = pa.u;                      // features 0..127
    rowq[32+lane] = pb.u;                      // features 128..255
  }
}

__global__ void init_kernel(unsigned* segmax, float* segsum, float* agg, int nl){
  int i = blockIdx.x*blockDim.x + threadIdx.x;
  if (i < nl){ segmax[i]=0u; segsum[i]=0.0f; }
  if (i < nl*128) agg[i]=0.0f;
}

// Pre-swizzle a (Ksrc x 128) f32 weight matrix into per-lane WMMA B fragments.
// dst[frag*512 + lane*16 + hh] with k=32s+8*(lane>>4)+(hh&7)+16*(hh>>3), n=16t+(lane&15).
__global__ void frag_kernel(const float* W, _Float16* dst, int Ksrc, int nS){
  int i = blockIdx.x*blockDim.x + threadIdx.x;
  int total = nS*8*512;
  if (i >= total) return;
  int hh = i & 15;
  int l  = (i>>4) & 31;
  int frag = i >> 9;
  int s = frag >> 3;
  int t = frag & 7;
  int k = 32*s + 8*(l>>4) + (hh&7) + ((hh>>3)<<4);
  int n = 16*t + (l&15);
  float v = (k < Ksrc) ? W[k*128 + n] : 0.0f;
  dst[i] = (_Float16)v;
}

__global__ __launch_bounds__(128) void logits_kernel(
    const float* h_prot, const float* h_lig,
    const float* ppos, const float* lpos,
    const int* edges,
    const _Float16* w1F, const float* b1, const float* w2, const float* b2p,
    float* logits, unsigned* segmax, int nTiles)
{
  __shared__ _Float16 catS[4][16*288];
  __shared__ int pidS[4][16];
  __shared__ int lidS[4][16];
  int widx = threadIdx.x >> 5;
  int lane = threadIdx.x & 31;
  const int* pidx = edges;
  const int* lidx = edges + NE;
  int nwaves = gridDim.x * 4;
  int gwave  = blockIdx.x*4 + widx;
  float b2 = b2p[0];
  for (int tile = gwave; tile < nTiles; tile += nwaves){
    if (lane == 0){                               // prefetch next tile's indices
      __builtin_prefetch(pidx + (tile+nwaves)*16, 0, 0);
      __builtin_prefetch(lidx + (tile+nwaves)*16, 0, 0);
    }
    float dloc = 0.0f;
    stage_geom(tile*16, lane, pidx, lidx, ppos, lpos, catS[widx], pidS[widx], lidS[widx], &dloc);
    WAVE_LDS_SYNC();
    stage_feat(lane, h_prot, h_lig, pidS[widx], lidS[widx], catS[widx]);
    WAVE_LDS_SYNC();
    {
      float accdot[8];
#pragma unroll
      for (int r=0;r<8;r++) accdot[r]=0.0f;
      int c0 = lane & 15;
      for (int t=0;t<8;t++){
        v8f acc = {};
#pragma unroll
        for (int s=0;s<9;s++){
          v16h a  = load_a_frag(catS[widx], lane, 32*s, 288);
          v16h bb = *(const v16h*)(w1F + (s*8+t)*512 + lane*16);
          acc = __builtin_amdgcn_wmma_f32_16x16x32_f16(false, a, false, bb, (short)0, acc, false, false);
        }
        int c = 16*t + c0;
        float bias = b1[c];
        float wc   = w2[c];
#pragma unroll
        for (int r=0;r<8;r++) accdot[r] += silu_f(acc[r]+bias)*wc;
      }
#pragma unroll
      for (int r=0;r<8;r++){
        float v = accdot[r];
        v += __shfl_xor(v,1); v += __shfl_xor(v,2);
        v += __shfl_xor(v,4); v += __shfl_xor(v,8);
        accdot[r]=v;
      }
      if ((lane&15)==0){
        int hi = lane>>4;
#pragma unroll
        for (int r=0;r<8;r++){
          int m = r + 8*hi;
          float lg = accdot[r] + b2;
          logits[tile*16+m] = lg;
          unsigned u = __float_as_uint(lg);
          u = (u & 0x80000000u) ? ~u : (u | 0x80000000u);   // order-preserving map
          atomicMax(segmax + lidS[widx][m], u);
        }
      }
    }
    WAVE_LDS_SYNC();                              // WAR fence before re-staging
  }
}

__global__ void denom_kernel(const int* lidx, const float* logits, const unsigned* segmax, float* segsum){
  int e = blockIdx.x*blockDim.x + threadIdx.x;
  if (e >= NE) return;
  int l = lidx[e];
  float mx = unmap_max(segmax[l]);
  atomicAdd(segsum + l, __expf(logits[e]-mx));
}

__global__ __launch_bounds__(128) void agg_kernel(
    const float* h_prot, const float* h_lig,
    const float* ppos, const float* lpos,
    const int* edges,
    const _Float16* w1F, const float* b1, const _Float16* w2F, const float* b2,
    const float* logits, const unsigned* segmax, const float* segsum,
    float* agg, int nTiles)
{
  __shared__ _Float16 catS[4][16*288];
  __shared__ _Float16 hidS[4][16*128];
  __shared__ float alphaS[4][16];
  __shared__ int pidS[4][16];
  __shared__ int lidS[4][16];
  int widx = threadIdx.x >> 5;
  int lane = threadIdx.x & 31;
  const int* pidx = edges;
  const int* lidx = edges + NE;
  int nwaves = gridDim.x * 4;
  int gwave  = blockIdx.x*4 + widx;
  for (int tile = gwave; tile < nTiles; tile += nwaves){
    if (lane == 0){
      __builtin_prefetch(pidx + (tile+nwaves)*16, 0, 0);
      __builtin_prefetch(lidx + (tile+nwaves)*16, 0, 0);
    }
    float dloc = 0.0f;
    stage_geom(tile*16, lane, pidx, lidx, ppos, lpos, catS[widx], pidS[widx], lidS[widx], &dloc);
    if (lane < 16){
      int e = tile*16 + lane;
      int l = lidx[e];
      float mx = unmap_max(segmax[l]);
      float w  = __expf(logits[e]-mx);
      alphaS[widx][lane] = w/(segsum[l]+1e-9f) * __expf(-dloc*dloc*(1.0f/32.0f)); // 2*sigma^2 = 32
    }
    WAVE_LDS_SYNC();
    stage_feat(lane, h_prot, h_lig, pidS[widx], lidS[widx], catS[widx]);
    WAVE_LDS_SYNC();
    {
      int c0 = lane & 15; int hi = lane>>4;
      for (int t=0;t<8;t++){           // hidden = silu(cat @ proj_w1 + b1)
        v8f acc = {};
#pragma unroll
        for (int s=0;s<9;s++){
          v16h a  = load_a_frag(catS[widx], lane, 32*s, 288);
          v16h bb = *(const v16h*)(w1F + (s*8+t)*512 + lane*16);
          acc = __builtin_amdgcn_wmma_f32_16x16x32_f16(false, a, false, bb, (short)0, acc, false, false);
        }
        int c = 16*t + c0;
        float bias = b1[c];
#pragma unroll
        for (int r=0;r<8;r++)
          hidS[widx][(r+8*hi)*128 + c] = (_Float16)silu_f(acc[r]+bias);
      }
    }
    WAVE_LDS_SYNC();
    {
      int c0 = lane & 15; int hi = lane>>4;
      for (int t=0;t<8;t++){           // v = hidden @ proj_w2 + b2, scaled by alpha
        v8f acc = {};
#pragma unroll
        for (int s=0;s<4;s++){
          v16h a  = load_a_frag(hidS[widx], lane, 32*s, 128);
          v16h bb = *(const v16h*)(w2F + (s*8+t)*512 + lane*16);
          acc = __builtin_amdgcn_wmma_f32_16x16x32_f16(false, a, false, bb, (short)0, acc, false, false);
        }
        int c = 16*t + c0;
        float bias = b2[c];
#pragma unroll
        for (int r=0;r<8;r++){
          int m = r + 8*hi;
          float val = (acc[r]+bias) * alphaS[widx][m];
          atomicAdd(agg + (size_t)lidS[widx][m]*128 + c, val);
        }
      }
    }
    WAVE_LDS_SYNC();                   // WAR fence before re-staging
  }
}

__global__ __launch_bounds__(128) void norm_kernel(const float* h_lig, const float* agg,
                                                   const float* gamma, const float* beta, float* out){
  int l = blockIdx.x; int c = threadIdx.x;
  float x = h_lig[(size_t)l*128 + c] + agg[(size_t)l*128 + c];
  float s = x, s2 = x*x;
#pragma unroll
  for (int m=1;m<32;m<<=1){ s += __shfl_xor(s,m); s2 += __shfl_xor(s2,m); }
  __shared__ float rs[4], rs2[4];
  int w = c >> 5;
  if ((c&31)==0){ rs[w]=s; rs2[w]=s2; }
  __syncthreads();
  float ts  = rs[0]+rs[1]+rs[2]+rs[3];
  float ts2 = rs2[0]+rs2[1]+rs2[2]+rs2[3];
  float mu  = ts * (1.0f/128.0f);
  float var = ts2 * (1.0f/128.0f) - mu*mu;
  out[(size_t)l*128 + c] = (x-mu)*rsqrtf(var+1e-5f)*gamma[c] + beta[c];
}

extern "C" void kernel_launch(void* const* d_in, const int* in_sizes, int n_in,
                              void* d_out, int out_size, void* d_ws, size_t ws_size,
                              hipStream_t stream)
{
  (void)in_sizes; (void)n_in; (void)out_size; (void)ws_size;
  const float* h_prot = (const float*)d_in[0];
  const float* h_lig  = (const float*)d_in[1];
  const float* ppos   = (const float*)d_in[2];
  const float* lpos   = (const float*)d_in[3];
  const int*   edges  = (const int*)  d_in[4];
  const float* att_w1 = (const float*)d_in[5];
  const float* att_b1 = (const float*)d_in[6];
  const float* att_w2 = (const float*)d_in[7];
  const float* att_b2 = (const float*)d_in[8];
  const float* proj_w1= (const float*)d_in[9];
  const float* proj_b1= (const float*)d_in[10];
  const float* proj_w2= (const float*)d_in[11];
  const float* proj_b2= (const float*)d_in[12];
  const float* gamma  = (const float*)d_in[13];
  const float* beta   = (const float*)d_in[14];

  char* ws = (char*)d_ws;
  float*     logits = (float*)    (ws + OFF_LOGITS);
  unsigned*  segmax = (unsigned*) (ws + OFF_SEGMAX);
  float*     segsum = (float*)    (ws + OFF_SEGSUM);
  float*     agg    = (float*)    (ws + OFF_AGG);
  _Float16*  w1aF   = (_Float16*) (ws + OFF_W1A);
  _Float16*  w1pF   = (_Float16*) (ws + OFF_W1P);
  _Float16*  w2pF   = (_Float16*) (ws + OFF_W2P);

  float* out_hp = (float*)d_out;
  float* out_hl = out_hp + (size_t)NP*128;

  // h_prot passes through unchanged
  hipMemcpyAsync(out_hp, h_prot, (size_t)NP*128*sizeof(float), hipMemcpyDeviceToDevice, stream);

  init_kernel<<<(NL*128+255)/256, 256, 0, stream>>>(segmax, segsum, agg, NL);
  frag_kernel<<<(9*8*512+255)/256, 256, 0, stream>>>(att_w1,  w1aF, INDIM, 9);
  frag_kernel<<<(9*8*512+255)/256, 256, 0, stream>>>(proj_w1, w1pF, INDIM, 9);
  frag_kernel<<<(4*8*512+255)/256, 256, 0, stream>>>(proj_w2, w2pF, HID,   4);

  logits_kernel<<<1024, 128, 0, stream>>>(h_prot, h_lig, ppos, lpos, edges,
                                          w1aF, att_b1, att_w2, att_b2,
                                          logits, segmax, NTILES);
  denom_kernel<<<(NE+255)/256, 256, 0, stream>>>(edges + NE, logits, segmax, segsum);
  agg_kernel<<<1024, 128, 0, stream>>>(h_prot, h_lig, ppos, lpos, edges,
                                       w1pF, proj_b1, w2pF, proj_b2,
                                       logits, segmax, segsum, agg, NTILES);
  norm_kernel<<<NL, 128, 0, stream>>>(h_lig, agg, gamma, beta, out_hl);
}
